// ResidualBlock_basic_37933151158491
// MI455X (gfx1250) — matrix operand order repaired and checked by reading, not compiled, not run on hardware
//
#include <hip/hip_runtime.h>

#define NN   600000
#define CC   64
#define KK3  27
#define MM   150000
#define NT   9375          // MM/16 tiles per kernel offset (exact)
#define TPW  5             // tiles per wave
#define WPB  8             // waves per 256-thread block
#define EPSF 1e-4f

typedef __attribute__((ext_vector_type(16))) __bf16 v16bf;
typedef __attribute__((ext_vector_type(4)))  __bf16 v4bf;
typedef __attribute__((ext_vector_type(8)))  float  v8f;

union V16U { uint4 u[2]; v16bf v; };

// ---------------------------------------------------------------------------
// Weight prepack: W[k][cin][cout] f32 -> bf16 in WMMA B-fragment order.
// Flat layout: (((k*2 + kk)*4 + n)*32 + lane)*16 + i
//   where cin = 32*kk + 8*(lane>>4) + (i&7) + 16*(i>>3), cout = 16*n + (lane&15)
// (16-bit B 32x16 layout: lanes 0-15 hold K=0-7,16-23, lanes 16-31 hold K=8-15,24-31)
// ---------------------------------------------------------------------------
__global__ void k_prepack(const float* __restrict__ W1, const float* __restrict__ W2,
                          __bf16* __restrict__ Wp1, __bf16* __restrict__ Wp2) {
  int tid = blockIdx.x * blockDim.x + threadIdx.x;
  if (tid >= KK3 * 4096) return;
  const float* W  = blockIdx.y ? W2  : W1;
  __bf16*      Wp = blockIdx.y ? Wp2 : Wp1;
  int i    = tid & 15;
  int lane = (tid >> 4) & 31;
  int n    = (tid >> 9) & 3;
  int kk   = (tid >> 11) & 1;
  int k    = tid >> 12;
  int cin  = 32 * kk + 8 * (lane >> 4) + (i & 7) + 16 * (i >> 3);
  int cout = 16 * n + (lane & 15);
  Wp[tid] = (__bf16)W[(k * CC + cin) * CC + cout];
}

// ---------------------------------------------------------------------------
// BN/ReLU #1 (f32 -> bf16) fused with zero-init of the f32 accumulator.
// ---------------------------------------------------------------------------
__global__ void k_bn1(const float* __restrict__ x,
                      const float* __restrict__ g, const float* __restrict__ b,
                      const float* __restrict__ m, const float* __restrict__ v,
                      __bf16* __restrict__ y, float* __restrict__ acc) {
  unsigned tid = blockIdx.x * blockDim.x + threadIdx.x;
  unsigned i = tid * 4u;
  if (i >= (unsigned)NN * CC) return;
  int c = (int)(i & 63u);
  float4 xv = *reinterpret_cast<const float4*>(x + i);
  float s0 = g[c + 0] * rsqrtf(v[c + 0] + EPSF);
  float s1 = g[c + 1] * rsqrtf(v[c + 1] + EPSF);
  float s2 = g[c + 2] * rsqrtf(v[c + 2] + EPSF);
  float s3 = g[c + 3] * rsqrtf(v[c + 3] + EPSF);
  v4bf o;
  o[0] = (__bf16)fmaxf((xv.x - m[c + 0]) * s0 + b[c + 0], 0.f);
  o[1] = (__bf16)fmaxf((xv.y - m[c + 1]) * s1 + b[c + 1], 0.f);
  o[2] = (__bf16)fmaxf((xv.z - m[c + 2]) * s2 + b[c + 2], 0.f);
  o[3] = (__bf16)fmaxf((xv.w - m[c + 3]) * s3 + b[c + 3], 0.f);
  *reinterpret_cast<v4bf*>(y + i) = o;
  float4 z = {0.f, 0.f, 0.f, 0.f};
  *reinterpret_cast<float4*>(acc + i) = z;
}

// ---------------------------------------------------------------------------
// BN/ReLU #2 (reads f32 conv1 accumulator -> bf16) fused with out = x,
// so conv2's atomic scatter-adds produce the residual sum directly.
// ---------------------------------------------------------------------------
__global__ void k_bn2(const float* __restrict__ acc1, const float* __restrict__ x,
                      const float* __restrict__ g, const float* __restrict__ b,
                      const float* __restrict__ m, const float* __restrict__ v,
                      __bf16* __restrict__ y, float* __restrict__ out) {
  unsigned tid = blockIdx.x * blockDim.x + threadIdx.x;
  unsigned i = tid * 4u;
  if (i >= (unsigned)NN * CC) return;
  int c = (int)(i & 63u);
  float4 av = *reinterpret_cast<const float4*>(acc1 + i);
  float4 xv = *reinterpret_cast<const float4*>(x + i);
  float s0 = g[c + 0] * rsqrtf(v[c + 0] + EPSF);
  float s1 = g[c + 1] * rsqrtf(v[c + 1] + EPSF);
  float s2 = g[c + 2] * rsqrtf(v[c + 2] + EPSF);
  float s3 = g[c + 3] * rsqrtf(v[c + 3] + EPSF);
  v4bf o;
  o[0] = (__bf16)fmaxf((av.x - m[c + 0]) * s0 + b[c + 0], 0.f);
  o[1] = (__bf16)fmaxf((av.y - m[c + 1]) * s1 + b[c + 1], 0.f);
  o[2] = (__bf16)fmaxf((av.z - m[c + 2]) * s2 + b[c + 2], 0.f);
  o[3] = (__bf16)fmaxf((av.w - m[c + 3]) * s3 + b[c + 3], 0.f);
  *reinterpret_cast<v4bf*>(y + i) = o;
  *reinterpret_cast<float4*>(out + i) = xv;   // residual init
}

// ---------------------------------------------------------------------------
// Rulebook conv: gather 16 rows (bf16) -> [16,64]@[64,64] via 8x
// v_wmma_f32_16x16x32_bf16 -> scatter-add f32 atomics.
// One wave per tile group; B fragments hoisted in VGPRs across TPW tiles.
// All global accesses use (scalar base + unsigned 32-bit byte offset) so the
// backend can select SADDR+voffset forms (no 64-bit per-lane address math).
// blockIdx.y = kernel offset k, blockIdx.x = tile group.
// ---------------------------------------------------------------------------
__global__ void __launch_bounds__(256)
k_conv(const __bf16* __restrict__ y, const __bf16* __restrict__ Wp,
       const int* __restrict__ in_idx, const int* __restrict__ out_idx,
       float* __restrict__ acc) {
  const unsigned lane = threadIdx.x & 31u;
  const unsigned wave = threadIdx.x >> 5;
  const unsigned k    = blockIdx.y;
  const unsigned half = lane >> 4;     // which 16-lane half
  const unsigned col  = lane & 15u;    // N column within a 16-wide tile

  // Hoist the 8 B fragments (kk=0..1 x n=0..3) for offset k.
  v16bf bfrag[8];
  {
    const char* wb = (const char*)Wp + (k * 8192u + lane * 32u);
    #pragma unroll
    for (int f = 0; f < 8; ++f) {
      V16U u;
      u.u[0] = *reinterpret_cast<const uint4*>(wb + f * 1024);
      u.u[1] = *reinterpret_cast<const uint4*>(wb + f * 1024 + 16);
      bfrag[f] = u.v;
    }
  }

  const int* ii = in_idx  + (size_t)k * MM;
  const int* oi = out_idx + (size_t)k * MM;

  const unsigned tile0 = (blockIdx.x * WPB + wave) * TPW;
  #pragma unroll 1
  for (unsigned t = 0; t < TPW; ++t) {
    const unsigned tile = tile0 + t;     // wave-uniform guard -> EXEC all 1s
    if (tile >= NT) break;
    const unsigned base = tile * 16u;

    // A fragment: 16-bit A 16x32 layout. Lane holds row tile*16+col,
    // K = {8*half..+7} U {16+8*half..+7} per 32-K slice (bf16, 16B chunks).
    const unsigned gi   = (unsigned)ii[base + col];
    const unsigned goff = gi * 128u + half * 16u;      // byte offset into y
    const char* rp = (const char*)y + goff;
    v16bf a0, a1;
    {
      V16U u;
      u.u[0] = *reinterpret_cast<const uint4*>(rp);
      u.u[1] = *reinterpret_cast<const uint4*>(rp + 32);
      a0 = u.v;
      u.u[0] = *reinterpret_cast<const uint4*>(rp + 64);
      u.u[1] = *reinterpret_cast<const uint4*>(rp + 96);
      a1 = u.v;
    }

    // Output row indices: 8 consecutive ints per half -> two b128 loads.
    int4 o0 = *reinterpret_cast<const int4*>(oi + base + 8u * half);
    int4 o1 = *reinterpret_cast<const int4*>(oi + base + 8u * half + 4);

    v8f c0 = {}, c1 = {}, c2 = {}, c3 = {};
    c0 = __builtin_amdgcn_wmma_f32_16x16x32_bf16(false, a0, false, bfrag[0], (short)0, c0, false, false);
    c1 = __builtin_amdgcn_wmma_f32_16x16x32_bf16(false, a0, false, bfrag[1], (short)0, c1, false, false);
    c2 = __builtin_amdgcn_wmma_f32_16x16x32_bf16(false, a0, false, bfrag[2], (short)0, c2, false, false);
    c3 = __builtin_amdgcn_wmma_f32_16x16x32_bf16(false, a0, false, bfrag[3], (short)0, c3, false, false);
    c0 = __builtin_amdgcn_wmma_f32_16x16x32_bf16(false, a1, false, bfrag[4], (short)0, c0, false, false);
    c1 = __builtin_amdgcn_wmma_f32_16x16x32_bf16(false, a1, false, bfrag[5], (short)0, c1, false, false);
    c2 = __builtin_amdgcn_wmma_f32_16x16x32_bf16(false, a1, false, bfrag[6], (short)0, c2, false, false);
    c3 = __builtin_amdgcn_wmma_f32_16x16x32_bf16(false, a1, false, bfrag[7], (short)0, c3, false, false);

    // Scatter-add: C/D layout -> VGPR r holds (M = r + 8*half, N = col).
    int orows[8] = {o0.x, o0.y, o0.z, o0.w, o1.x, o1.y, o1.z, o1.w};
    #pragma unroll
    for (int r = 0; r < 8; ++r) {
      const unsigned ooff = (unsigned)orows[r] * 256u + col * 4u;  // bytes
      float* dst = (float*)((char*)acc + ooff);
      atomicAdd(dst +  0, c0[r]);
      atomicAdd(dst + 16, c1[r]);
      atomicAdd(dst + 32, c2[r]);
      atomicAdd(dst + 48, c3[r]);
    }
  }
}

// ---------------------------------------------------------------------------
extern "C" void kernel_launch(void* const* d_in, const int* in_sizes, int n_in,
                              void* d_out, int out_size, void* d_ws, size_t ws_size,
                              hipStream_t stream) {
  const float* x  = (const float*)d_in[0];
  const float* W1 = (const float*)d_in[1];
  const float* W2 = (const float*)d_in[2];
  const float* g1 = (const float*)d_in[3];
  const float* b1 = (const float*)d_in[4];
  const float* m1 = (const float*)d_in[5];
  const float* v1 = (const float*)d_in[6];
  const float* g2 = (const float*)d_in[7];
  const float* b2 = (const float*)d_in[8];
  const float* m2 = (const float*)d_in[9];
  const float* v2 = (const float*)d_in[10];
  const int* in_idx  = (const int*)d_in[11];
  const int* out_idx = (const int*)d_in[12];
  float* out = (float*)d_out;

  char* ws = (char*)d_ws;
  float*  acc  = (float*)ws;                                // [N,64] f32 accumulator
  __bf16* ybuf = (__bf16*)(ws + (size_t)NN * CC * 4);       // [N,64] bf16 activations
  __bf16* Wp1  = (__bf16*)(ws + (size_t)NN * CC * 6);       // 27*4096 bf16
  __bf16* Wp2  = Wp1 + (size_t)KK3 * 4096;

  const int ew = (NN * CC) / 4;
  const dim3 cg((NT + WPB * TPW - 1) / (WPB * TPW), KK3);

  // 1) pack both weight sets to bf16 B-fragment layout
  k_prepack<<<dim3((KK3 * 4096 + 255) / 256, 2), 256, 0, stream>>>(W1, W2, Wp1, Wp2);
  // 2) y = bnrelu1(x) (bf16); acc = 0
  k_bn1<<<(ew + 255) / 256, 256, 0, stream>>>(x, g1, b1, m1, v1, ybuf, acc);
  // 3) acc += subconv(y, W1)
  k_conv<<<cg, 256, 0, stream>>>(ybuf, Wp1, in_idx, out_idx, acc);
  // 4) y = bnrelu2(acc) (bf16); out = x (residual init)
  k_bn2<<<(ew + 255) / 256, 256, 0, stream>>>(acc, x, g2, b2, m2, v2, ybuf, out);
  // 5) out += subconv(y, W2)  -> out = x + conv2(bnrelu2(conv1(bnrelu1(x))))
  k_conv<<<cg, 256, 0, stream>>>(ybuf, Wp2, in_idx, out_idx, out);
}
